// ODEModel_30313879175818
// MI455X (gfx1250) — matrix operand order repaired and checked by reading, not compiled, not run on hardware
//
#include <hip/hip_runtime.h>

// ---------------------------------------------------------------------------
// MI455X (gfx1250) implementation.
//  Kernel 1: mean over T (memory bound, ~315MB stream -> ~14us at 23.3 TB/s).
//  Kernel 2: RK4 ODE loop; MLP evals on the matrix pipe via
//            v_wmma_f32_16x16x32_f16 (f32 accumulate). Weights resident in
//            VGPRs (B-layout) for all 101 steps; activations transposed
//            through per-wave LDS (bank-conflict-padded). Biases folded into
//            the GEMMs via an augmented constant-1 K-channel. Epilogue is
//            single-op v_max_num ReLU (inline asm, avoids the NaN-canonicalize
//            pair) + v_cvt_pk_rtz_f16_f32 + b16/b16_d16_hi stores.
// ---------------------------------------------------------------------------

typedef __attribute__((ext_vector_type(16))) _Float16 v16h;
typedef __attribute__((ext_vector_type(8)))  _Float16 v8h;
typedef __attribute__((ext_vector_type(2)))  __fp16   v2hf;  // cvt_pkrtz result
typedef __attribute__((ext_vector_type(8)))  float    v8f;

#define WMMA16(A, Bm, C) \
    __builtin_amdgcn_wmma_f32_16x16x32_f16(false, (A), false, (Bm), (short)0, (C), false, false)

// one-instruction ReLU; inline asm so LLVM cannot re-insert the canonicalize
// (input comes from a WMMA accumulator, never a signaling NaN)
__device__ __forceinline__ float relu1(float x) {
    float r;
    asm("v_max_num_f32 %0, %1, 0" : "=v"(r) : "v"(x));
    return r;
}

// pack two f32 -> two f16 in one v_cvt_pk_rtz_f16_f32, reinterpret as _Float16
union pk2 {
    v2hf v;
    _Float16 e[2];
};
__device__ __forceinline__ pk2 pkcvt(float a, float b) {
    pk2 p;
    p.v = __builtin_amdgcn_cvt_pkrtz(a, b);
    return p;
}

// Fixed reference shapes (padded): D=6 -> K1P=32, H=50 -> HP=64, out D=6 -> DP=16
#define DIN   6
#define HID   50
#define BCH   50      // bias channel column in hidden pad region (HID..63)
#define K1P   32
#define HP    64
#define HSTR  72      // padded LDS row stride (halfs): 144B, 16B-aligned,
                      // rows 8 apart land 32 banks offset -> no half-wave clash
#define DP    16

// ---------------------------------------------------------------------------
// Kernel 1: y0[b][d] = mean_t x0[b][t][d]
// ---------------------------------------------------------------------------
__global__ __launch_bounds__(256) void mean_kernel(const float* __restrict__ x0,
                                                   float* __restrict__ y0,
                                                   int B, int T, int D) {
    long i = (long)blockIdx.x * blockDim.x + threadIdx.x;   // over B*D
    long total = (long)B * D;
    if (i >= total) return;
    long b = i / D;
    int  d = (int)(i - b * D);
    const float* p = x0 + b * (long)T * D + d;
    float s = 0.f;
    for (int t = 0; t < T; ++t) s += p[(long)t * D];
    y0[i] = s * (1.0f / (float)T);
}

// ---------------------------------------------------------------------------
// Kernel 2: RK4 with WMMA MLP. One wave = 16 batch rows. 8 waves / block.
// ---------------------------------------------------------------------------
__global__ __launch_bounds__(256) void ode_kernel(const float* __restrict__ y0g,
                                                  const float* __restrict__ tspan,
                                                  const float* __restrict__ W1,
                                                  const float* __restrict__ b1,
                                                  const float* __restrict__ W2,
                                                  const float* __restrict__ b2,
                                                  const float* __restrict__ W3,
                                                  const float* __restrict__ b3,
                                                  float* __restrict__ out,
                                                  int B, int nsteps) {
    // Weight staging (transposed [N][K], zero padded, biases folded in).
    __shared__ __attribute__((aligned(32))) _Float16 sW1T[HP * K1P];   // [64][32]
    __shared__ __attribute__((aligned(32))) _Float16 sW2T[HP * HP];    // [64][64]
    __shared__ __attribute__((aligned(32))) _Float16 sW3T[DP * HP];    // [16][64]
    // Per-wave activation staging (C-layout -> A-layout transpose buffers).
    __shared__ __attribute__((aligned(32))) _Float16 sY[8][16 * K1P];  // 1KB/wave
    __shared__ __attribute__((aligned(32))) _Float16 sH[8][16 * HSTR]; // 2.25KB/wave

    const int tid = threadIdx.x;

    // ---- stage weights with folded biases ----
    // A1 has a constant-1 column at k=6 (set once in sY below).
    // Hidden activations carry a constant-1 "bias channel" at column n=BCH.
    for (int i = tid; i < HP * K1P; i += 256) {
        int n = i >> 5, k = i & 31;
        float w = 0.f;
        if (n < HID) {
            if (k < DIN)       w = W1[k * HID + n];
            else if (k == DIN) w = b1[n];            // + b1 via A col k=6 == 1
        } else if (n == BCH && k == DIN) {
            w = 1.f;                                  // H1[.,BCH] = 1
        }
        sW1T[i] = (_Float16)w;
    }
    for (int i = tid; i < HP * HP; i += 256) {
        int n = i >> 6, k = i & 63;
        float w = 0.f;
        if (n < HID) {
            if (k < HID)       w = W2[k * HID + n];
            else if (k == BCH) w = b2[n];            // + b2 via H1 col BCH == 1
        } else if (n == BCH && k == BCH) {
            w = 1.f;                                  // H2[.,BCH] = 1
        }
        sW2T[i] = (_Float16)w;
    }
    for (int i = tid; i < DP * HP; i += 256) {
        int n = i >> 6, k = i & 63;
        float w = 0.f;
        if (n < DIN) {
            if (k < HID)       w = W3[k * DIN + n];
            else if (k == BCH) w = b3[n];            // + b3 via H2 col BCH == 1
        }
        sW3T[i] = (_Float16)w;
    }
    __syncthreads();

    const int wv     = tid >> 5;        // wave id in block (wave32)
    const int lane   = tid & 31;
    const int nn     = lane & 15;       // C/B layout: column index within tile
    const int m0     = (lane >> 4) * 8; // C layout: row base (0 or 8)
    const int mA     = lane & 15;       // A layout: row = lane%16
    const int kA     = (lane >> 4) * 16;// A layout: K half base (0 or 16)
    const int kb     = (lane >> 4) * 16;// B layout: K half base (0 or 16)
    _Float16* Yb = sY[wv];
    _Float16* Hb = sH[wv];

    // zero the layer-1 A staging pad, then plant the constant-1 bias column
    // at k=6 (eval stores only touch k<6, so both survive the whole loop)
    for (int i = lane; i < 16 * K1P; i += 32) Yb[i] = (_Float16)0.f;
    if (lane < 16) Yb[lane * K1P + DIN] = (_Float16)1.f;

    // ---- load resident B-operand weight tiles (kept in VGPRs all 101 steps) ----
    v16h B1[4], B2[2][4], B3[2];
#pragma unroll
    for (int t = 0; t < 4; ++t)
        B1[t] = *(const v16h*)&sW1T[(t * 16 + nn) * K1P + kb];
#pragma unroll
    for (int kc = 0; kc < 2; ++kc)
#pragma unroll
        for (int t = 0; t < 4; ++t)
            B2[kc][t] = *(const v16h*)&sW2T[(t * 16 + nn) * HP + kc * 32 + kb];
#pragma unroll
    for (int kc = 0; kc < 2; ++kc)
        B3[kc] = *(const v16h*)&sW3T[nn * HP + kc * 32 + kb];

    // ---- load y0 tile in C layout (VGPR r <-> row m0+r, col nn) ----
    const long rowBase = ((long)blockIdx.x * 8 + wv) * 16;
    float yv[8];
#pragma unroll
    for (int r = 0; r < 8; ++r) {
        long row = rowBase + m0 + r;
        yv[r] = (nn < DIN && row < B) ? y0g[row * DIN + nn] : 0.f;
    }

    // ---- one MLP evaluation: kout = f(yt), both in C-layout f32 regs ----
    auto evalf = [&](const float* yt, float* kout) {
        // C-layout y -> LDS [16][32] f16 (A staging), valid dims only.
        // Packed cvt: rows (r, r+1) in one v_cvt_pk_rtz_f16_f32.
        if (nn < DIN) {
#pragma unroll
            for (int r = 0; r < 8; r += 2) {
                pk2 p = pkcvt(yt[r], yt[r + 1]);
                Yb[(m0 + r)     * K1P + nn] = p.e[0];
                Yb[(m0 + r + 1) * K1P + nn] = p.e[1];  // ds_store_b16_d16_hi
            }
        }
        // A tile (16x32): two contiguous 8-K chunks per lane
        union { v16h v; v8h h[2]; } A1;
        A1.h[0] = *(const v8h*)&Yb[mA * K1P + kA];
        A1.h[1] = *(const v8h*)&Yb[mA * K1P + kA + 8];

        // layer 1 (+b1 folded): [16x32]x[32x64] -> 4 WMMAs
        v8f acc[4];
#pragma unroll
        for (int t = 0; t < 4; ++t) {
            v8f c = {};
            acc[t] = WMMA16(A1.v, B1[t], c);
        }
        // ReLU (1-op v_max) + packed cvt + store to H staging [16][HSTR]
#pragma unroll
        for (int t = 0; t < 4; ++t)
#pragma unroll
            for (int r = 0; r < 8; r += 2) {
                pk2 p = pkcvt(relu1(acc[t][r]), relu1(acc[t][r + 1]));
                Hb[(m0 + r)     * HSTR + t * 16 + nn] = p.e[0];
                Hb[(m0 + r + 1) * HSTR + t * 16 + nn] = p.e[1];
            }
        // layer 2 (+b2 folded): A = H1 (two 16x32 A tiles), 8 WMMAs
        union { v16h v; v8h h[2]; } A2[2];
#pragma unroll
        for (int kc = 0; kc < 2; ++kc) {
            A2[kc].h[0] = *(const v8h*)&Hb[mA * HSTR + kc * 32 + kA];
            A2[kc].h[1] = *(const v8h*)&Hb[mA * HSTR + kc * 32 + kA + 8];
        }
#pragma unroll
        for (int t = 0; t < 4; ++t) {
            v8f c = {};
            c = WMMA16(A2[0].v, B2[0][t], c);
            c = WMMA16(A2[1].v, B2[1][t], c);
            acc[t] = c;
        }
#pragma unroll
        for (int t = 0; t < 4; ++t)
#pragma unroll
            for (int r = 0; r < 8; r += 2) {
                pk2 p = pkcvt(relu1(acc[t][r]), relu1(acc[t][r + 1]));
                Hb[(m0 + r)     * HSTR + t * 16 + nn] = p.e[0];
                Hb[(m0 + r + 1) * HSTR + t * 16 + nn] = p.e[1];
            }
        // layer 3 (+b3 folded): [16xHP]x[HPx16] -> 2 WMMAs, D is kout directly
        union { v16h v; v8h h[2]; } A3[2];
#pragma unroll
        for (int kc = 0; kc < 2; ++kc) {
            A3[kc].h[0] = *(const v8h*)&Hb[mA * HSTR + kc * 32 + kA];
            A3[kc].h[1] = *(const v8h*)&Hb[mA * HSTR + kc * 32 + kA + 8];
        }
        v8f c3 = {};
        c3 = WMMA16(A3[0].v, B3[0], c3);
        c3 = WMMA16(A3[1].v, B3[1], c3);
#pragma unroll
        for (int r = 0; r < 8; ++r)
            kout[r] = c3[r];     // pad cols n>=6: zero weights, no bias -> 0
    };

    // ---- RK4 loop (f32 state/accumulation throughout) ----
    for (int s = 0; s < nsteps; ++s) {
        const float dt = tspan[s + 1] - tspan[s];
        float k[8], yt[8], accm[8];
        evalf(yv, k);                                         // k1
#pragma unroll
        for (int r = 0; r < 8; ++r) {
            accm[r] = yv[r] + (dt * (1.f / 6.f)) * k[r];
            yt[r]   = yv[r] + 0.5f * dt * k[r];
        }
        evalf(yt, k);                                         // k2
#pragma unroll
        for (int r = 0; r < 8; ++r) {
            accm[r] += (dt * (1.f / 3.f)) * k[r];
            yt[r]    = yv[r] + 0.5f * dt * k[r];
        }
        evalf(yt, k);                                         // k3
#pragma unroll
        for (int r = 0; r < 8; ++r) {
            accm[r] += (dt * (1.f / 3.f)) * k[r];
            yt[r]    = yv[r] + dt * k[r];
        }
        evalf(yt, k);                                         // k4
#pragma unroll
        for (int r = 0; r < 8; ++r)
            yv[r] = accm[r] + (dt * (1.f / 6.f)) * k[r];
    }

    // ---- store yT ----
    if (nn < DIN) {
#pragma unroll
        for (int r = 0; r < 8; ++r) {
            long row = rowBase + m0 + r;
            if (row < B) out[row * DIN + nn] = yv[r];
        }
    }
}

// ---------------------------------------------------------------------------
extern "C" void kernel_launch(void* const* d_in, const int* in_sizes, int n_in,
                              void* d_out, int out_size, void* d_ws, size_t ws_size,
                              hipStream_t stream) {
    const float* x0    = (const float*)d_in[0];
    const float* tspan = (const float*)d_in[1];
    const float* W1    = (const float*)d_in[2];
    const float* b1    = (const float*)d_in[3];
    const float* W2    = (const float*)d_in[4];
    const float* b2    = (const float*)d_in[5];
    const float* W3    = (const float*)d_in[6];
    const float* b3    = (const float*)d_in[7];
    float* out = (float*)d_out;

    const int D      = in_sizes[7];              // 6
    const int B      = out_size / D;             // 65536
    const int T      = in_sizes[0] / (B * D);    // 200
    const int nsteps = in_sizes[1] - 1;          // 101

    float* y0 = (float*)d_ws;                    // B*D floats of scratch

    const long nbd = (long)B * D;
    mean_kernel<<<(int)((nbd + 255) / 256), 256, 0, stream>>>(x0, y0, B, T, D);

    const int blocks = (B + 127) / 128;          // 128 rows / block (8 waves x 16)
    ode_kernel<<<blocks, 256, 0, stream>>>(y0, tspan, W1, b1, W2, b2, W3, b3,
                                           out, B, nsteps);
}
